// M11_81071802679817
// MI455X (gfx1250) — matrix operand order repaired and checked by reading, not compiled, not run on hardware
//
#include <hip/hip_runtime.h>
#include <hip/hip_bf16.h>

typedef __attribute__((ext_vector_type(16))) _Float16 v16h;
typedef __attribute__((ext_vector_type(8)))  float    v8f;

#define N_NODES 100000
#define N_EDGES 1600000
#define N_GRAPH 2000
#define F_IN    40
#define FE_DIM  16
#define H_DIM   64
#define HM_DIM  128
#define NPG     50          // nodes per graph (N/G)
#define NEG_SLOPE 0.01f
#define BN_EPS  1e-5f

#define NB64    ((N_NODES + 63) / 64)     // 1563 row-blocks
#define NPAD    (NB64 * 64)               // 100032 padded rows

static_assert(N_EDGES % 64 == 0, "edge grid must tile exactly");

__device__ __forceinline__ float leaky_f(float x) { return x >= 0.f ? x : NEG_SLOPE * x; }

// ---------------------------------------------------------------------------
// Utility kernels
// ---------------------------------------------------------------------------
__global__ void k_zero4(float4* __restrict__ p, int n4) {
    int i = blockIdx.x * blockDim.x + threadIdx.x;
    if (i < n4) p[i] = make_float4(0.f, 0.f, 0.f, 0.f);
}

// W[K, Ncol] (f32, row major) -> Wt[Ncol, Kpad] (f16), zero-padded in K.
__global__ void k_wprep(const float* __restrict__ W, _Float16* __restrict__ Wt,
                        int K, int Ncol, int Kpad) {
    int i = blockIdx.x * blockDim.x + threadIdx.x;
    if (i >= Ncol * Kpad) return;
    int col = i / Kpad, k = i % Kpad;
    Wt[i] = (k < K) ? (_Float16)W[k * Ncol + col] : (_Float16)0.f;
}

// ---------------------------------------------------------------------------
// WMMA fragment loaders (CDNA5 V_WMMA_F32_16X16X32_F16 layouts, wave32)
// A (16x32 f16): lane<16 -> row=lane, K in {0..7, 16..23}; lane>=16 -> row=lane-16,
//                K in {8..15, 24..31}
// ---------------------------------------------------------------------------
__device__ __forceinline__ v16h load_A_frag(const _Float16* tile, int ldh, int lane, int kstep) {
    int row = lane & 15;
    int kb  = (lane & 16) ? 8 : 0;
    const _Float16* p = tile + row * ldh + kstep * 32 + kb;
    v16h a;
#pragma unroll
    for (int j = 0; j < 8; ++j) { a[j] = p[j]; a[8 + j] = p[16 + j]; }
    return a;
}

// B (32x16 f16): lane<16 -> col=lane, K in {0..7, 16..23}; lane>=16 -> col=lane-16, K in {8..15, 24..31}
__device__ __forceinline__ v16h load_B_frag(const _Float16* __restrict__ Wt, int Kpad,
                                            int col_base, int lane, int kstep) {
    int col = col_base + (lane & 15);
    int kb  = kstep * 32 + ((lane & 16) ? 8 : 0);
    const _Float16* p = Wt + col * Kpad + kb;
    v16h b;
#pragma unroll
    for (int j = 0; j < 8; ++j) { b[j] = p[j]; b[8 + j] = p[16 + j]; }
    return b;
}

__device__ __forceinline__ void st_h4(_Float16* d, float a, float b, float c, float e) {
    d[0] = (_Float16)a; d[1] = (_Float16)b; d[2] = (_Float16)c; d[3] = (_Float16)e;
}

// ---------------------------------------------------------------------------
// GINE edge stage: agg[dst] += relu(x[src] + edge_attr @ We + be)
// 256 threads / block, exactly 64 edges / block.
// ---------------------------------------------------------------------------
template <int CI>
__global__ void k_edge(const float* __restrict__ x, int xstride,
                       const int* __restrict__ ei,      // [2, E]
                       const float* __restrict__ ea,    // [E, 16]
                       const float* __restrict__ We,    // [16, CI]
                       const float* __restrict__ be,    // [CI]
                       float* __restrict__ agg)         // [NPAD, 64]
{
    __shared__ float sWe[FE_DIM][H_DIM];
    __shared__ float sbe[H_DIM];
    __shared__ float sea[64][FE_DIM];
    __shared__ int   ssrc[64], sdst[64];

    const int t = threadIdx.x;
    for (int idx = t; idx < FE_DIM * CI; idx += 256)
        sWe[idx / CI][idx % CI] = We[idx];
    if (t < CI) sbe[t] = be[t];

    const int base = blockIdx.x * 64;
    {   // 64 edges * 4 float4 = 256 vector loads -> one per thread
        int el = t >> 2, k4 = (t & 3) * 4;
        *(float4*)&sea[el][k4] = *(const float4*)(ea + (size_t)(base + el) * FE_DIM + k4);
    }
    if (t < 64) {
        ssrc[t] = ei[base + t];
        sdst[t] = ei[N_EDGES + base + t];
    }
    __syncthreads();

    const int c   = t & 63;
    const int el0 = t >> 6;   // 0..3
    if (c < CI) {
#pragma unroll 4
        for (int p = 0; p < 16; ++p) {
            int el = p * 4 + el0;
            float v = sbe[c];
#pragma unroll
            for (int k = 0; k < FE_DIM; ++k) v = fmaf(sea[el][k], sWe[k][c], v);
            v += x[(size_t)ssrc[el] * xstride + c];
            v = fmaxf(v, 0.f);
            atomicAdd(&agg[(size_t)sdst[el] * H_DIM + c], v);
        }
    }
}

// ---------------------------------------------------------------------------
// y1 = ((1+eps)*x + agg) @ W1 + b1        (WMMA, K padded to 64)
// + per-column sum / sumsq accumulation for training-mode BatchNorm.
// Output rows are padded (NPAD); invalid rows store 0 -> stats stay exact.
// ---------------------------------------------------------------------------
__global__ void k_lin1(const float* __restrict__ x, int xstride, int ci,
                       const float* __restrict__ agg,
                       const float* __restrict__ epsp,
                       const _Float16* __restrict__ W1t,   // [64][64] f16
                       const float* __restrict__ b1,
                       float* __restrict__ y1,             // [NPAD,64]
                       float* __restrict__ gsum, float* __restrict__ gsumsq)
{
    constexpr int LDH = 72;                    // 144B rows -> conflict-free, 16B aligned
    __shared__ _Float16 sh[64 * LDH];
    __shared__ float ssum[H_DIM], ssq[H_DIM];

    const int t = threadIdx.x, wave = t >> 5, lane = t & 31;
    const int row0 = blockIdx.x * 64;
    const float eps1 = 1.0f + epsp[0];

    if (t < H_DIM) { ssum[t] = 0.f; ssq[t] = 0.f; }

    // Build the 64x64 f16 A tile (x may be the unpadded kernel input -> guard).
    for (int idx = t; idx < 64 * 16; idx += 128) {
        int r = idx >> 4, c4 = (idx & 15) * 4;
        int gr = row0 + r;
        float4 xv = make_float4(0.f, 0.f, 0.f, 0.f), av = xv;
        if (gr < N_NODES && c4 < ci) {
            xv = *(const float4*)(x + (size_t)gr * xstride + c4);
            av = *(const float4*)(agg + (size_t)gr * H_DIM + c4);
        }
        st_h4(sh + r * LDH + c4, fmaf(eps1, xv.x, av.x), fmaf(eps1, xv.y, av.y),
                                 fmaf(eps1, xv.z, av.z), fmaf(eps1, xv.w, av.w));
    }
    __syncthreads();

    const _Float16* atile = sh + (wave * 16) * LDH;
    v16h A0 = load_A_frag(atile, LDH, lane, 0);
    v16h A1 = load_A_frag(atile, LDH, lane, 1);
    const int mb    = (lane & 16) ? 8 : 0;
    const int rbase = row0 + wave * 16 + mb;

#pragma unroll
    for (int tc = 0; tc < 4; ++tc) {
        v16h B0 = load_B_frag(W1t, H_DIM, tc * 16, lane, 0);
        v16h B1 = load_B_frag(W1t, H_DIM, tc * 16, lane, 1);
        v8f acc = {};
        acc = __builtin_amdgcn_wmma_f32_16x16x32_f16(false, A0, false, B0, (short)0, acc, false, false);
        acc = __builtin_amdgcn_wmma_f32_16x16x32_f16(false, A1, false, B1, (short)0, acc, false, false);
        int col = tc * 16 + (lane & 15);
        float bias = b1[col];
        float ps = 0.f, pq = 0.f;
#pragma unroll
        for (int r = 0; r < 8; ++r) {
            float v = (rbase + r < N_NODES) ? acc[r] + bias : 0.f;   // select, no branch
            y1[(size_t)(rbase + r) * H_DIM + col] = v;               // padded store
            ps += v; pq += v * v;
        }
        atomicAdd(&ssum[col], ps);
        atomicAdd(&ssq[col], pq);
    }
    __syncthreads();
    if (t < H_DIM) {
        atomicAdd(&gsum[t], ssum[t]);
        atomicAdd(&gsumsq[t], ssq[t]);
    }
}

// BN finalize: scale = g * rsqrt(var + eps); shift = b - mean * scale
__global__ void k_bnfin(const float* __restrict__ sum, const float* __restrict__ sumsq,
                        const float* __restrict__ g, const float* __restrict__ b,
                        float* __restrict__ scale, float* __restrict__ shift, float nInv)
{
    int t = threadIdx.x;
    if (t < H_DIM) {
        float m  = sum[t] * nInv;
        float v  = sumsq[t] * nInv - m * m;
        float rs = rsqrtf(v + BN_EPS);
        float sc = g[t] * rs;
        scale[t] = sc;
        shift[t] = b[t] - m * sc;
    }
}

// ---------------------------------------------------------------------------
// h2 = leaky(bn1(y1)) @ W2 + b2   (WMMA)   + outer-BN stats
// y1 is padded -> unconditional tile build; padded output rows store 0.
// ---------------------------------------------------------------------------
__global__ void k_lin2(const float* __restrict__ y1,
                       const float* __restrict__ scale1, const float* __restrict__ shift1,
                       const _Float16* __restrict__ W2t,   // [64][64] f16
                       const float* __restrict__ b2,
                       float* __restrict__ h2,             // [NPAD,64]
                       float* __restrict__ gsum, float* __restrict__ gsumsq)
{
    constexpr int LDH = 72;
    __shared__ _Float16 sh[64 * LDH];
    __shared__ float ssum[H_DIM], ssq[H_DIM];

    const int t = threadIdx.x, wave = t >> 5, lane = t & 31;
    const int row0 = blockIdx.x * 64;

    if (t < H_DIM) { ssum[t] = 0.f; ssq[t] = 0.f; }

    for (int idx = t; idx < 64 * 16; idx += 128) {
        int r = idx >> 4, c4 = (idx & 15) * 4;
        int gr = row0 + r;
        float4 y = *(const float4*)(y1 + (size_t)gr * H_DIM + c4);
        st_h4(sh + r * LDH + c4,
              leaky_f(fmaf(y.x, scale1[c4 + 0], shift1[c4 + 0])),
              leaky_f(fmaf(y.y, scale1[c4 + 1], shift1[c4 + 1])),
              leaky_f(fmaf(y.z, scale1[c4 + 2], shift1[c4 + 2])),
              leaky_f(fmaf(y.w, scale1[c4 + 3], shift1[c4 + 3])));
    }
    __syncthreads();

    const _Float16* atile = sh + (wave * 16) * LDH;
    v16h A0 = load_A_frag(atile, LDH, lane, 0);
    v16h A1 = load_A_frag(atile, LDH, lane, 1);
    const int mb    = (lane & 16) ? 8 : 0;
    const int rbase = row0 + wave * 16 + mb;

#pragma unroll
    for (int tc = 0; tc < 4; ++tc) {
        v16h B0 = load_B_frag(W2t, H_DIM, tc * 16, lane, 0);
        v16h B1 = load_B_frag(W2t, H_DIM, tc * 16, lane, 1);
        v8f acc = {};
        acc = __builtin_amdgcn_wmma_f32_16x16x32_f16(false, A0, false, B0, (short)0, acc, false, false);
        acc = __builtin_amdgcn_wmma_f32_16x16x32_f16(false, A1, false, B1, (short)0, acc, false, false);
        int col = tc * 16 + (lane & 15);
        float bias = b2[col];
        float ps = 0.f, pq = 0.f;
#pragma unroll
        for (int r = 0; r < 8; ++r) {
            float v = (rbase + r < N_NODES) ? acc[r] + bias : 0.f;
            h2[(size_t)(rbase + r) * H_DIM + col] = v;
            ps += v; pq += v * v;
        }
        atomicAdd(&ssum[col], ps);
        atomicAdd(&ssq[col], pq);
    }
    __syncthreads();
    if (t < H_DIM) {
        atomicAdd(&gsum[t], ssum[t]);
        atomicAdd(&gsumsq[t], ssq[t]);
    }
}

// x = leaky(bn2(h2))   elementwise over padded rows, float4
__global__ void k_bnact4(const float4* __restrict__ h2, const float* __restrict__ scale,
                         const float* __restrict__ shift, float4* __restrict__ xout)
{
    int i = blockIdx.x * blockDim.x + threadIdx.x;
    if (i < NPAD * H_DIM / 4) {
        int c4 = (i & 15) * 4;
        float4 v = h2[i];
        float4 o;
        o.x = leaky_f(fmaf(v.x, scale[c4 + 0], shift[c4 + 0]));
        o.y = leaky_f(fmaf(v.y, scale[c4 + 1], shift[c4 + 1]));
        o.z = leaky_f(fmaf(v.z, scale[c4 + 2], shift[c4 + 2]));
        o.w = leaky_f(fmaf(v.w, scale[c4 + 3], shift[c4 + 3]));
        xout[i] = o;
    }
}

// global_add_pool with contiguous segments: batch[n] = n / 50
__global__ void k_pool(const float* __restrict__ x, float* __restrict__ pool)
{
    int g = blockIdx.x, c = threadIdx.x;   // 64 threads
    const float* p = x + (size_t)g * NPG * H_DIM + c;
    float s = 0.f;
#pragma unroll
    for (int i = 0; i < NPG; ++i) s += p[i * H_DIM];
    pool[(size_t)g * H_DIM + c] = s;
}

// ---------------------------------------------------------------------------
// mlp0: out = leaky(concat(x, pool[batch]) @ W + b)     [NPAD,128] (WMMA K=128)
// ---------------------------------------------------------------------------
__global__ void k_mlp0(const float* __restrict__ x, const float* __restrict__ pool,
                       const int* __restrict__ batch,
                       const _Float16* __restrict__ Wt,    // [128][128] f16
                       const float* __restrict__ b,
                       float* __restrict__ out)            // [NPAD,128]
{
    constexpr int LDH = 136;                 // 272B rows -> conflict-free, 16B aligned
    __shared__ _Float16 sh[64 * LDH];
    const int t = threadIdx.x, wave = t >> 5, lane = t & 31;
    const int row0 = blockIdx.x * 64;

    for (int idx = t; idx < 64 * 32; idx += 128) {
        int r = idx >> 5, c4 = (idx & 31) * 4;
        int gr = row0 + r;
        int grc = (gr < N_NODES) ? gr : (N_NODES - 1);   // batch[] is unpadded input
        float4 v = (c4 < H_DIM)
            ? *(const float4*)(x + (size_t)gr * H_DIM + c4)                       // padded
            : *(const float4*)(pool + (size_t)batch[grc] * H_DIM + (c4 - H_DIM));
        st_h4(sh + r * LDH + c4, v.x, v.y, v.z, v.w);
    }
    __syncthreads();

    const _Float16* atile = sh + (wave * 16) * LDH;
    v16h A[4];
#pragma unroll
    for (int s = 0; s < 4; ++s) A[s] = load_A_frag(atile, LDH, lane, s);
    const int mb    = (lane & 16) ? 8 : 0;
    const int rbase = row0 + wave * 16 + mb;

#pragma unroll
    for (int tc = 0; tc < 8; ++tc) {
        v8f acc = {};
#pragma unroll
        for (int s = 0; s < 4; ++s) {
            v16h B = load_B_frag(Wt, HM_DIM, tc * 16, lane, s);
            acc = __builtin_amdgcn_wmma_f32_16x16x32_f16(false, A[s], false, B, (short)0, acc, false, false);
        }
        int col = tc * 16 + (lane & 15);
        float bias = b[col];
#pragma unroll
        for (int r = 0; r < 8; ++r) {
            float v = (rbase + r < N_NODES) ? leaky_f(acc[r] + bias) : 0.f;
            out[(size_t)(rbase + r) * HM_DIM + col] = v;
        }
    }
}

// mlp1: out = leaky(in @ W + b); safe in-place (each block only touches its rows)
__global__ void k_mlp1(const float* in,
                       const _Float16* __restrict__ Wt,    // [128][128] f16
                       const float* __restrict__ b,
                       float* out)                         // may alias `in`; padded
{
    constexpr int LDH = 136;
    __shared__ _Float16 sh[64 * LDH];
    const int t = threadIdx.x, wave = t >> 5, lane = t & 31;
    const int row0 = blockIdx.x * 64;

    for (int idx = t; idx < 64 * 32; idx += 128) {
        int r = idx >> 5, c4 = (idx & 31) * 4;
        float4 v = *(const float4*)(in + (size_t)(row0 + r) * HM_DIM + c4);  // padded
        st_h4(sh + r * LDH + c4, v.x, v.y, v.z, v.w);
    }
    __syncthreads();

    const _Float16* atile = sh + (wave * 16) * LDH;
    v16h A[4];
#pragma unroll
    for (int s = 0; s < 4; ++s) A[s] = load_A_frag(atile, LDH, lane, s);
    const int mb    = (lane & 16) ? 8 : 0;
    const int rbase = row0 + wave * 16 + mb;

#pragma unroll
    for (int tc = 0; tc < 8; ++tc) {
        v8f acc = {};
#pragma unroll
        for (int s = 0; s < 4; ++s) {
            v16h B = load_B_frag(Wt, HM_DIM, tc * 16, lane, s);
            acc = __builtin_amdgcn_wmma_f32_16x16x32_f16(false, A[s], false, B, (short)0, acc, false, false);
        }
        int col = tc * 16 + (lane & 15);
        float bias = b[col];
#pragma unroll
        for (int r = 0; r < 8; ++r) {
            float v = (rbase + r < N_NODES) ? leaky_f(acc[r] + bias) : 0.f;
            out[(size_t)(rbase + r) * HM_DIM + col] = v;
        }
    }
}

// final: out[n, 0..1] = m1[n] @ Wf + bf   (tiny K=128, C=2; LDS-tiled scalar)
// d_out is harness-owned and unpadded -> keep the row guard here only.
__global__ void k_final(const float* __restrict__ m1, const float* __restrict__ Wf,
                        const float* __restrict__ bf, float* __restrict__ out)
{
    __shared__ float sW[HM_DIM * 2];
    __shared__ float srow[64][HM_DIM];
    const int t = threadIdx.x;   // 256
    for (int i = t; i < HM_DIM * 2; i += 256) sW[i] = Wf[i];
    const int row0 = blockIdx.x * 64;
    {
        int r = t >> 2, c4 = (t & 3) * 32;   // 64 rows * 128 cols in 8 float4/thread
#pragma unroll
        for (int j = 0; j < 8; ++j) {
            int c = c4 + j * 4;
            *(float4*)&srow[r][c] = *(const float4*)(m1 + (size_t)(row0 + r) * HM_DIM + c); // padded
        }
    }
    __syncthreads();
    if (t < 128) {
        int r = t >> 1, cls = t & 1;
        int gr = row0 + r;
        if (gr < N_NODES) {
            float s = bf[cls];
#pragma unroll
            for (int k = 0; k < HM_DIM; ++k) s = fmaf(srow[r][k], sW[k * 2 + cls], s);
            out[(size_t)gr * 2 + cls] = s;
        }
    }
}

// ---------------------------------------------------------------------------
// Host launcher
// ---------------------------------------------------------------------------
extern "C" void kernel_launch(void* const* d_in, const int* in_sizes, int n_in,
                              void* d_out, int out_size, void* d_ws, size_t ws_size,
                              hipStream_t stream)
{
    (void)in_sizes; (void)n_in; (void)out_size; (void)ws_size;

    const float* x_in  = (const float*)d_in[0];
    const int*   ei    = (const int*)d_in[1];      // [2, E]
    const float* ea    = (const float*)d_in[2];    // [E, 16]
    const int*   batch = (const int*)d_in[3];      // [N]
    auto cp = [&](int layer, int j) { return (const float*)d_in[4 + layer * 11 + j]; };
    const float* mlp0_W  = (const float*)d_in[37];
    const float* mlp0_b  = (const float*)d_in[38];
    const float* mlp1_W  = (const float*)d_in[39];
    const float* mlp1_b  = (const float*)d_in[40];
    const float* final_W = (const float*)d_in[41];
    const float* final_b = (const float*)d_in[42];

    // Workspace carve-up (floats); all row buffers padded to NPAD rows
    float* fws = (float*)d_ws;
    size_t o = 0;
    float* xbuf  = fws + o; o += (size_t)NPAD * H_DIM;   // current node features
    float* aggb  = fws + o; o += (size_t)NPAD * H_DIM;   // agg / h2 / (m0 lower half)
    float* y1b   = fws + o; o += (size_t)NPAD * H_DIM;   // y1 / (m0 upper half)
    float* m0    = aggb;                                 // [NPAD,128] spans aggb..y1b
    float* poolb = fws + o; o += (size_t)N_GRAPH * H_DIM;
    float* stats = fws + o; o += 6 * 256;                // per-BN: sum|sumsq|scale|shift
    _Float16* wt = (_Float16*)(fws + o);
    _Float16* W1t[3], * W2t[3];
    for (int i = 0; i < 3; ++i) { W1t[i] = wt + i * 4096; W2t[i] = wt + 12288 + i * 4096; }
    _Float16* m0t = wt + 24576;        // 128*128
    _Float16* m1t = m0t + 16384;       // 128*128

    const int NBE = N_EDGES / 64;                     // 25000 (exact)
    const int NZ4 = (NPAD * H_DIM / 4 + 255) / 256;   // padded elementwise grid

    k_zero4<<<2, 256, 0, stream>>>((float4*)stats, 6 * 64);
    for (int i = 0; i < 3; ++i) {
        k_wprep<<<16, 256, 0, stream>>>(cp(i, 3), W1t[i], (i == 0 ? F_IN : H_DIM), H_DIM, H_DIM);
        k_wprep<<<16, 256, 0, stream>>>(cp(i, 7), W2t[i], H_DIM, H_DIM, H_DIM);
    }
    k_wprep<<<64, 256, 0, stream>>>(mlp0_W, m0t, HM_DIM, HM_DIM, HM_DIM);
    k_wprep<<<64, 256, 0, stream>>>(mlp1_W, m1t, HM_DIM, HM_DIM, HM_DIM);

    const float* xcur = x_in;
    int xs = F_IN, ci = F_IN;
    for (int i = 0; i < 3; ++i) {
        float* s0 = stats + (2 * i) * 256;     float* q0 = s0 + 64; float* sc0 = s0 + 128; float* sh0 = s0 + 192;
        float* s1 = stats + (2 * i + 1) * 256; float* q1 = s1 + 64; float* sc1 = s1 + 128; float* sh1 = s1 + 192;

        k_zero4<<<NZ4, 256, 0, stream>>>((float4*)aggb, NPAD * H_DIM / 4);
        if (ci == F_IN)
            k_edge<F_IN><<<NBE, 256, 0, stream>>>(xcur, xs, ei, ea, cp(i, 1), cp(i, 2), aggb);
        else
            k_edge<H_DIM><<<NBE, 256, 0, stream>>>(xcur, xs, ei, ea, cp(i, 1), cp(i, 2), aggb);

        k_lin1<<<NB64, 128, 0, stream>>>(xcur, xs, ci, aggb, cp(i, 0), W1t[i], cp(i, 4), y1b, s0, q0);
        k_bnfin<<<1, 64, 0, stream>>>(s0, q0, cp(i, 5), cp(i, 6), sc0, sh0, 1.0f / N_NODES);
        k_lin2<<<NB64, 128, 0, stream>>>(y1b, sc0, sh0, W2t[i], cp(i, 8), aggb, s1, q1);
        k_bnfin<<<1, 64, 0, stream>>>(s1, q1, cp(i, 9), cp(i, 10), sc1, sh1, 1.0f / N_NODES);
        k_bnact4<<<NZ4, 256, 0, stream>>>((const float4*)aggb, sc1, sh1, (float4*)xbuf);

        xcur = xbuf; xs = H_DIM; ci = H_DIM;
    }

    k_pool<<<N_GRAPH, 64, 0, stream>>>(xbuf, poolb);
    k_mlp0<<<NB64, 128, 0, stream>>>(xbuf, poolb, batch, m0t, mlp0_b, m0);
    k_mlp1<<<NB64, 128, 0, stream>>>(m0, m1t, mlp1_b, m0);   // safe in place
    k_final<<<NB64, 256, 0, stream>>>(m0, final_W, final_b, (float*)d_out);
}
